// MemeplexActorCritic_13477607375213
// MI455X (gfx1250) — matrix operand order repaired and checked by reading, not compiled, not run on hardware
//
#include <hip/hip_runtime.h>
#include <hip/hip_bf16.h>

// MI455X / gfx1250, wave32. One workgroup (8 waves) per batch element; full
// pipeline fused in LDS; all GEMMs via v_wmma_f32_16x16x32_f16.
// Workspace layout (halves): weight matrices pre-packed into B-fragment order
// by pack_weight_f16 (total 73,728 halves = 147,456 bytes).

typedef __attribute__((ext_vector_type(16))) _Float16 v16h;
typedef __attribute__((ext_vector_type(8)))  _Float16 v8h;
typedef __attribute__((ext_vector_type(8)))  float    v8f;
typedef __attribute__((ext_vector_type(4)))  float    v4f;

#define N_AG   64
#define OBS_D  128
#define HID_D  128
#define COMM_D 16
#define MEME_D 16
#define NMEMES 8
#define NACT   20

struct Params {
  const float* obs;
  const float* b_e1; const float* b_e2;
  const float* b_s1; const float* b_s2;
  const float* meme;
  const float* b_k; const float* b_v; const float* b_q;
  const float* b_a1; const float* b_a2;
  const _Float16 *We1, *We2, *Ws1, *Ws2, *Wq, *Wk, *Wv, *Wa1, *Wa2;
  float* out;
};

// ---------------------------------------------------------------------------
// Pack a fp32 weight matrix (Krows x Ncols, row-major) into WMMA B-fragment
// order, zero-padded to 32xK tiles / 16xN tiles. Fragment layout (per the
// CDNA5 ISA 16-bit B 32x16 striping): lane L holds column N = L%16; kb=(L<16?0:8);
// elements 0-7 -> K=kb+e, 8-15 -> K=16+kb+(e-8). One 32B blob per lane.
// ---------------------------------------------------------------------------
__global__ void pack_weight_f16(const float* __restrict__ W, int Krows, int Ncols,
                                int Ntiles, _Float16* __restrict__ dst) {
  const int lane = threadIdx.x;                 // 32 threads per block
  const int kt = blockIdx.x / Ntiles;
  const int nt = blockIdx.x % Ntiles;
  const int n  = nt * 16 + (lane & 15);
  const int k0 = kt * 32 + ((lane < 16) ? 0 : 8);
  _Float16* d = dst + ((size_t)blockIdx.x * 32 + lane) * 16;
#pragma unroll
  for (int e = 0; e < 16; ++e) {
    int k = k0 + (e < 8 ? e : 8 + e);
    float v = (k < Krows && n < Ncols) ? W[(size_t)k * Ncols + n] : 0.f;
    d[e] = (_Float16)v;
  }
}

// ---------------------------------------------------------------------------
// Generic 64xN GEMM: A (f16, LDS, row-major, stride lda halves) x Wfrag
// (pre-packed B fragments in global). 4 M-tiles x Ntiles N-tiles round-robined
// over 8 waves; the t%Ntiles mapping keeps nt fixed per wave so B fragments
// are loop-invariant and live in registers. A-fragment per the ISA 16-bit A
// 16x32 layout: lane L holds row M=L%16; K-base kb=(L<16?0:8); elements
// 0-7 -> K=kb+e, 8-15 -> K=16+kb+(e-8)  => two 16B LDS loads per fragment.
// ---------------------------------------------------------------------------
template <typename StoreFn>
__device__ __forceinline__ void wmma_gemm(const _Float16* As, int lda,
                                          const _Float16* Wfrag, int Ntiles,
                                          int Ksteps, StoreFn store) {
  const int lane = threadIdx.x & 31;
  const int wave = threadIdx.x >> 5;
  for (int t = wave; t < 4 * Ntiles; t += 8) {
    const int mt = t / Ntiles;
    const int nt = t % Ntiles;
    v8f acc = {};
    const _Float16* arow = As + (mt * 16 + (lane & 15)) * lda + ((lane < 16) ? 0 : 8);
    for (int kt = 0; kt < Ksteps; ++kt) {
      v8h lo = *(const v8h*)(arow + kt * 32);
      v8h hi = *(const v8h*)(arow + kt * 32 + 16);
      v16h a = __builtin_shufflevector(lo, hi, 0,1,2,3,4,5,6,7,8,9,10,11,12,13,14,15);
      v16h bm = *(const v16h*)(Wfrag + ((size_t)(kt * Ntiles + nt) * 32 + lane) * 16);
      acc = __builtin_amdgcn_wmma_f32_16x16x32_f16(false, a, false, bm,
                                                   (short)0, acc, false, false);
    }
    store(mt, nt, lane, acc);
  }
}

// ---------------------------------------------------------------------------
// Fused pipeline: one block per batch. LDS: sX/sY 64x160 f16, sA 64x64 f32.
// ---------------------------------------------------------------------------
__global__ void __launch_bounds__(256) memeplex_fused(Params p) {
  __shared__ _Float16 sX[N_AG * 160];
  __shared__ _Float16 sY[N_AG * 160];
  __shared__ float    sA[N_AG * N_AG];
  float* sYf = (float*)sY;                      // alias: 80 f32 per 320B row

  const int b    = blockIdx.x;
  const int tid  = threadIdx.x;
  const int lane = tid & 31;
  const int wave = tid >> 5;

  // ---- stage 0: obs tile -> f16 in sX[,0:128)  (vectorized b128 loads) ------
  const float* obs = p.obs + (size_t)b * N_AG * OBS_D;
  for (int o4 = tid; o4 < (N_AG * OBS_D) / 4; o4 += 256) {
    v4f v = *(const v4f*)(obs + o4 * 4);
    int i = o4 >> 5, c = (o4 & 31) * 4;         // row, col
    _Float16* d = sX + i * 160 + c;
    d[0] = (_Float16)v.x; d[1] = (_Float16)v.y;
    d[2] = (_Float16)v.z; d[3] = (_Float16)v.w;
  }
  __syncthreads();

  // ---- encoder: h = relu(relu(obs@We1+b)@We2+b) ------------------------------
  wmma_gemm(sX, 160, p.We1, 8, 4, [&](int mt, int nt, int ln, v8f acc) {
    int n = ln & 15, col = nt * 16 + n, m0 = mt * 16 + ((ln < 16) ? 0 : 8);
    float bv = p.b_e1[col];
#pragma unroll
    for (int r = 0; r < 8; ++r) {
      float v = acc[r] + bv; v = v > 0.f ? v : 0.f;
      sY[(m0 + r) * 160 + col] = (_Float16)v;
    }
  });
  __syncthreads();
  wmma_gemm(sY, 160, p.We2, 8, 4, [&](int mt, int nt, int ln, v8f acc) {
    int n = ln & 15, col = nt * 16 + n, m0 = mt * 16 + ((ln < 16) ? 0 : 8);
    float bv = p.b_e2[col];
#pragma unroll
    for (int r = 0; r < 8; ++r) {
      float v = acc[r] + bv; v = v > 0.f ? v : 0.f;
      sX[(m0 + r) * 160 + col] = (_Float16)v;          // h in sX[,0:128)
    }
  });
  __syncthreads();

  // ---- selector: softmax(relu(h@Ws1+b)@Ws2+b) -------------------------------
  wmma_gemm(sX, 160, p.Ws1, 4, 4, [&](int mt, int nt, int ln, v8f acc) {
    int n = ln & 15, col = nt * 16 + n, m0 = mt * 16 + ((ln < 16) ? 0 : 8);
    float bv = p.b_s1[col];
#pragma unroll
    for (int r = 0; r < 8; ++r) {
      float v = acc[r] + bv; v = v > 0.f ? v : 0.f;
      sY[(m0 + r) * 160 + col] = (_Float16)v;
    }
  });
  __syncthreads();
  wmma_gemm(sY, 160, p.Ws2, 1, 2, [&](int mt, int nt, int ln, v8f acc) {
    int n = ln & 15, m0 = mt * 16 + ((ln < 16) ? 0 : 8);
    float bv = (n < NMEMES) ? p.b_s2[n] : 0.f;
#pragma unroll
    for (int r = 0; r < 8; ++r) sA[(m0 + r) * 16 + n] = acc[r] + bv;  // 64x16 staging
  });
  __syncthreads();
  if (tid < N_AG) {                             // softmax over 8 memes per row
    float mx = -1e30f;
#pragma unroll
    for (int m = 0; m < NMEMES; ++m) mx = fmaxf(mx, sA[tid * 16 + m]);
    float e[NMEMES], s = 0.f;
#pragma unroll
    for (int m = 0; m < NMEMES; ++m) { e[m] = __expf(sA[tid * 16 + m] - mx); s += e[m]; }
    float inv = 1.f / s;
#pragma unroll
    for (int m = 0; m < NMEMES; ++m) sA[tid * 16 + m] = e[m] * inv;
  }
  __syncthreads();

  // ---- active memes -> sX[,128:144); zero sX[,144:160) for padded K/V K-dim --
  for (int o = tid; o < N_AG * MEME_D; o += 256) {
    int i = o >> 4, d = o & 15;
    const float* mp = p.meme + ((size_t)i * NMEMES) * MEME_D + d;
    float a = 0.f;
#pragma unroll
    for (int m = 0; m < NMEMES; ++m) a += sA[i * 16 + m] * mp[m * MEME_D];
    sX[i * 160 + 128 + d] = (_Float16)a;
    sX[i * 160 + 144 + d] = (_Float16)0.f;
  }
  __syncthreads();

  // ---- Q/K/V heads (12 WMMA tile jobs over 8 waves) -> f32 alias of sY ------
  // row layout in sYf (pitch 80): K[0:16) V[16:32) Q[32:48)
  for (int t = wave; t < 12; t += 8) {
    int which = t >> 2, mt = t & 3;
    const _Float16* Wf; const float* bias; int ks, c0;
    if (which == 0)      { Wf = p.Wk; bias = p.b_k; ks = 5; c0 = 0;  }
    else if (which == 1) { Wf = p.Wv; bias = p.b_v; ks = 5; c0 = 16; }
    else                 { Wf = p.Wq; bias = p.b_q; ks = 4; c0 = 32; }
    v8f acc = {};
    const _Float16* arow = sX + (mt * 16 + (lane & 15)) * 160 + ((lane < 16) ? 0 : 8);
    for (int kt = 0; kt < ks; ++kt) {
      v8h lo = *(const v8h*)(arow + kt * 32);
      v8h hi = *(const v8h*)(arow + kt * 32 + 16);
      v16h a = __builtin_shufflevector(lo, hi, 0,1,2,3,4,5,6,7,8,9,10,11,12,13,14,15);
      v16h bm = *(const v16h*)(Wf + ((size_t)kt * 32 + lane) * 16);
      acc = __builtin_amdgcn_wmma_f32_16x16x32_f16(false, a, false, bm,
                                                   (short)0, acc, false, false);
    }
    int n = lane & 15, m0 = mt * 16 + ((lane < 16) ? 0 : 8);
    float bv = bias[n];
#pragma unroll
    for (int r = 0; r < 8; ++r) sYf[(m0 + r) * 80 + c0 + n] = acc[r] + bv;
  }
  __syncthreads();

  // ---- attention scores: Q.K^T / 4 (VALU; 16-wide dots) ----------------------
  for (int o = tid; o < N_AG * N_AG; o += 256) {
    int i = o >> 6, j = o & 63;
    const float* q = sYf + i * 80 + 32;
    const float* k = sYf + j * 80;
    float d = 0.f;
#pragma unroll
    for (int c = 0; c < COMM_D; ++c) d += q[c] * k[c];
    sA[o] = d * 0.25f;
  }
  __syncthreads();
  if (tid < N_AG) {                             // self-masked softmax per row
    float mx = -1e30f;
    for (int j = 0; j < N_AG; ++j) if (j != tid) mx = fmaxf(mx, sA[tid * 64 + j]);
    float s = 0.f;
    for (int j = 0; j < N_AG; ++j) {
      float e = (j == tid) ? 0.f : __expf(sA[tid * 64 + j] - mx);
      sA[tid * 64 + j] = e; s += e;
    }
    float inv = (s > 0.f) ? (1.f / s) : 0.f;
    for (int j = 0; j < N_AG; ++j) sA[tid * 64 + j] *= inv;
  }
  __syncthreads();

  // ---- context = attn @ V -> sX[,144:160); sX now holds feat (160 wide) ------
  for (int o = tid; o < N_AG * COMM_D; o += 256) {
    int i = o >> 4, c = o & 15;
    float a = 0.f;
    for (int j = 0; j < N_AG; ++j) a += sA[i * 64 + j] * sYf[j * 80 + 16 + c];
    sX[i * 160 + 144 + c] = (_Float16)a;
  }
  __syncthreads();

  // ---- actor: logits = relu(feat@Wa1+b)@Wa2+b --------------------------------
  wmma_gemm(sX, 160, p.Wa1, 8, 5, [&](int mt, int nt, int ln, v8f acc) {
    int n = ln & 15, col = nt * 16 + n, m0 = mt * 16 + ((ln < 16) ? 0 : 8);
    float bv = p.b_a1[col];
#pragma unroll
    for (int r = 0; r < 8; ++r) {
      float v = acc[r] + bv; v = v > 0.f ? v : 0.f;
      sY[(m0 + r) * 160 + col] = (_Float16)v;
    }
  });
  __syncthreads();
  float* out = p.out + (size_t)b * N_AG * NACT;
  wmma_gemm(sY, 160, p.Wa2, 2, 4, [&](int mt, int nt, int ln, v8f acc) {
    int n = ln & 15, col = nt * 16 + n, m0 = mt * 16 + ((ln < 16) ? 0 : 8);
    if (col < NACT) {
      float bv = p.b_a2[col];
#pragma unroll
      for (int r = 0; r < 8; ++r) out[(size_t)(m0 + r) * NACT + col] = acc[r] + bv;
    }
  });
}

// ---------------------------------------------------------------------------
extern "C" void kernel_launch(void* const* d_in, const int* in_sizes, int n_in,
                              void* d_out, int out_size, void* d_ws, size_t ws_size,
                              hipStream_t stream) {
  (void)in_sizes; (void)n_in; (void)out_size; (void)ws_size;

  _Float16* ws  = (_Float16*)d_ws;              // needs 147,456 B of workspace
  _Float16* We1 = ws;                           // 128x128
  _Float16* We2 = We1 + 128 * 128;              // 128x128
  _Float16* Ws1 = We2 + 128 * 128;              // 128x64
  _Float16* Ws2 = Ws1 + 128 * 64;               // 64x16 (N 8->16)
  _Float16* Wq  = Ws2 + 64 * 16;                // 128x16
  _Float16* Wk  = Wq  + 128 * 16;               // 160x16 (K 144->160)
  _Float16* Wv  = Wk  + 160 * 16;               // 160x16
  _Float16* Wa1 = Wv  + 160 * 16;               // 160x128
  _Float16* Wa2 = Wa1 + 160 * 128;              // 128x32 (N 20->32)

  pack_weight_f16<<<4 * 8, 32, 0, stream>>>((const float*)d_in[1], 128, 128, 8, We1);
  pack_weight_f16<<<4 * 8, 32, 0, stream>>>((const float*)d_in[3], 128, 128, 8, We2);
  pack_weight_f16<<<4 * 4, 32, 0, stream>>>((const float*)d_in[5], 128, 64, 4, Ws1);
  pack_weight_f16<<<2 * 1, 32, 0, stream>>>((const float*)d_in[7], 64, 8, 1, Ws2);
  pack_weight_f16<<<5 * 1, 32, 0, stream>>>((const float*)d_in[10], 144, 16, 1, Wk);
  pack_weight_f16<<<5 * 1, 32, 0, stream>>>((const float*)d_in[12], 144, 16, 1, Wv);
  pack_weight_f16<<<4 * 1, 32, 0, stream>>>((const float*)d_in[14], 128, 16, 1, Wq);
  pack_weight_f16<<<5 * 8, 32, 0, stream>>>((const float*)d_in[16], 160, 128, 8, Wa1);
  pack_weight_f16<<<4 * 2, 32, 0, stream>>>((const float*)d_in[18], 128, 20, 2, Wa2);

  Params p;
  p.obs  = (const float*)d_in[0];
  p.b_e1 = (const float*)d_in[2];  p.b_e2 = (const float*)d_in[4];
  p.b_s1 = (const float*)d_in[6];  p.b_s2 = (const float*)d_in[8];
  p.meme = (const float*)d_in[9];
  p.b_k  = (const float*)d_in[11]; p.b_v  = (const float*)d_in[13];
  p.b_q  = (const float*)d_in[15];
  p.b_a1 = (const float*)d_in[17]; p.b_a2 = (const float*)d_in[19];
  p.We1 = We1; p.We2 = We2; p.Ws1 = Ws1; p.Ws2 = Ws2;
  p.Wq = Wq; p.Wk = Wk; p.Wv = Wv; p.Wa1 = Wa1; p.Wa2 = Wa2;
  p.out = (float*)d_out;

  memeplex_fused<<<4096, 256, 0, stream>>>(p);
}
